// PatchDecoder_84318797955409
// MI455X (gfx1250) — matrix operand order repaired and checked by reading, not compiled, not run on hardware
//
#include <hip/hip_runtime.h>
#include <hip/hip_bf16.h>

typedef __attribute__((ext_vector_type(16))) _Float16 v16h;
typedef __attribute__((ext_vector_type(8)))  _Float16 v8h;
typedef __attribute__((ext_vector_type(8)))  float    v8f;
typedef __attribute__((ext_vector_type(4)))  float    v4f;

#define HID  128
#define ZD   32
#define NB   5
#define ROWP 136   // padded LDS row stride in halfs (272B) -> conflict-free A-frag loads

// ---------------------------------------------------------------------------
// D = A(16x32 f16) * B(32x16 f16) + C(16x16 f32)
__device__ __forceinline__ v8f wmma16x16x32(v16h a, v16h b, v8f c) {
  return __builtin_amdgcn_wmma_f32_16x16x32_f16(false, a, false, b, (short)0, c,
                                                false, false);
}

// A-fragment (contiguous k order): lane holds row m, element e -> K=32*kk+16*halfSel+e.
// kbase = 32*kk + 8*halfSel.
__device__ __forceinline__ v16h load_a_frag(const _Float16* actrow, int kbase) {
  v8h lo = *(const v8h*)(actrow + kbase + (kbase & 8));
  v8h hi = *(const v8h*)(actrow + kbase + (kbase & 8) + 8);
  return __builtin_shufflevector(lo, hi, 0,1,2,3,4,5,6,7,8,9,10,11,12,13,14,15);
}

// ---------------------------------------------------------------------------
__global__ void cvt_f32_f16_kernel(const float* __restrict__ src,
                                   _Float16* __restrict__ dst, int n) {
  int i = blockIdx.x * 256 + threadIdx.x;
  if (i < n) dst[i] = (_Float16)src[i];
}

// ---------------------------------------------------------------------------
__global__ __launch_bounds__(256) void PatchDecoder_84318797955409_kernel(
    const float* __restrict__ zf,  const float* __restrict__ qry,
    const _Float16* __restrict__ Wp, const _Float16* __restrict__ Wc,
    const _Float16* __restrict__ W0, const _Float16* __restrict__ W1,
    const float* __restrict__ bp, const float* __restrict__ bc,
    const float* __restrict__ b0, const float* __restrict__ b1,
    const float* __restrict__ wout, const float* __restrict__ bout,
    float* __restrict__ out, int nq, const int* __restrict__ gsize)
{
  __shared__ __align__(16) _Float16 lds_act[8 * 16 * ROWP];

  const int tid  = threadIdx.x;
  const int lane = tid & 31;
  const int wave = tid >> 5;
  const int hs   = lane >> 4;     // half-of-wave select
  const int lm   = lane & 15;     // A row m / B,C column n (mod 16)
  const int tile = blockIdx.x * 8 + wave;
  const int ntiles = (nq + 15) >> 4;
  if (tile >= ntiles) return;     // wave-uniform: EXEC stays all-ones for WMMA
  const int G = *gsize;

  _Float16* act    = &lds_act[wave * (16 * ROWP)];
  _Float16* actrow = act + lm * ROWP;

  // ----------------- trilinear gather-interpolate (latent: 16x32) ----------
  const int q  = tile * 16 + lm;
  const int qc = q < nq ? q : nq - 1;
  const float qx = qry[qc * 3 + 0];
  const float qy = qry[qc * 3 + 1];
  const float qz = qry[qc * 3 + 2];
  const float fx = floorf(qx), fy = floorf(qy), fz = floorf(qz);
  const float rx = qx - fx, ry = qy - fy, rz = qz - fz;
  const int bx = (int)fx, by = (int)fy, bz = (int)fz;

  v4f s0 = {0.f, 0.f, 0.f, 0.f}, s1 = s0, s2 = s0, s3 = s0;
  #pragma unroll
  for (int c = 0; c < 8; ++c) {
    const int ox = (c >> 2) & 1, oy = (c >> 1) & 1, oz = c & 1;
    const float w = (ox ? rx : 1.f - rx) * (oy ? ry : 1.f - ry)
                  * (oz ? rz : 1.f - rz);
    const size_t flat = (size_t)(((bx + ox) * G + (by + oy)) * G + (bz + oz));
    const v4f* fp = (const v4f*)(zf + flat * ZD + 16 * hs);
    s0 += w * fp[0]; s1 += w * fp[1]; s2 += w * fp[2]; s3 += w * fp[3];
  }
  // stage latent as f16 row [m][k], this lane covers k = 16*hs .. 16*hs+15
  v8h p0, p1;
  #pragma unroll
  for (int j = 0; j < 4; ++j) {
    p0[j]     = (_Float16)s0[j];  p0[4 + j] = (_Float16)s1[j];
    p1[j]     = (_Float16)s2[j];  p1[4 + j] = (_Float16)s3[j];
  }
  *(v8h*)(actrow + 16 * hs)     = p0;
  *(v8h*)(actrow + 16 * hs + 8) = p1;

  const v16h latA = load_a_frag(actrow, 8 * hs);   // 16x32 latent A-matrix

  // ----------------- fc_p : net = latent @ Wp^T + bp -----------------------
  v8f acc[8];
  #pragma unroll
  for (int nt = 0; nt < 8; ++nt) {
    const int n = 16 * nt + lm;
    const v16h b = *(const v16h*)(Wp + n * ZD + 16 * hs);
    v8f cz = {0.f, 0.f, 0.f, 0.f, 0.f, 0.f, 0.f, 0.f};
    v8f d = wmma16x16x32(latA, b, cz);
    const float bia = bp[n];
    #pragma unroll
    for (int j = 0; j < 8; ++j) d[j] += bia;
    acc[nt] = d;
  }

  // ----------------- 5 ResNet blocks ---------------------------------------
  for (int i = 0; i < NB; ++i) {
    const _Float16* Wci = Wc + i * HID * ZD;
    const _Float16* W0i = W0 + i * HID * HID;
    const _Float16* W1i = W1 + i * HID * HID;

    // net += latent @ Wc[i]^T + bc[i]
    #pragma unroll
    for (int nt = 0; nt < 8; ++nt) {
      const int n = 16 * nt + lm;
      const v16h b = *(const v16h*)(Wci + n * ZD + 16 * hs);
      v8f d = wmma16x16x32(latA, b, acc[nt]);
      const float bia = bc[i * HID + n];
      #pragma unroll
      for (int j = 0; j < 8; ++j) d[j] += bia;
      acc[nt] = d;
    }

    // stage relu(net) -> LDS  (C-layout: lane holds n=lm, rows r+8*hs)
    #pragma unroll
    for (int nt = 0; nt < 8; ++nt) {
      #pragma unroll
      for (int r = 0; r < 8; ++r)
        act[(r + 8 * hs) * ROWP + 16 * nt + lm] =
            (_Float16)fmaxf(acc[nt][r], 0.f);
    }
    v16h a0 = load_a_frag(actrow,  0 + 8 * hs);
    v16h a1 = load_a_frag(actrow, 32 + 8 * hs);
    v16h a2 = load_a_frag(actrow, 64 + 8 * hs);
    v16h a3 = load_a_frag(actrow, 96 + 8 * hs);

    // h = relu(net) @ W0[i]^T + b0[i]
    v8f h[8];
    #pragma unroll
    for (int nt = 0; nt < 8; ++nt) {
      const int n = 16 * nt + lm;
      const float bia = b0[i * HID + n];
      v8f cc;
      #pragma unroll
      for (int j = 0; j < 8; ++j) cc[j] = bia;
      cc = wmma16x16x32(a0, *(const v16h*)(W0i + n * HID +  0 + 16 * hs), cc);
      cc = wmma16x16x32(a1, *(const v16h*)(W0i + n * HID + 32 + 16 * hs), cc);
      cc = wmma16x16x32(a2, *(const v16h*)(W0i + n * HID + 64 + 16 * hs), cc);
      cc = wmma16x16x32(a3, *(const v16h*)(W0i + n * HID + 96 + 16 * hs), cc);
      h[nt] = cc;
    }

    // stage relu(h) -> LDS
    #pragma unroll
    for (int nt = 0; nt < 8; ++nt) {
      #pragma unroll
      for (int r = 0; r < 8; ++r)
        act[(r + 8 * hs) * ROWP + 16 * nt + lm] =
            (_Float16)fmaxf(h[nt][r], 0.f);
    }
    a0 = load_a_frag(actrow,  0 + 8 * hs);
    a1 = load_a_frag(actrow, 32 + 8 * hs);
    a2 = load_a_frag(actrow, 64 + 8 * hs);
    a3 = load_a_frag(actrow, 96 + 8 * hs);

    // net += relu(h) @ W1[i]^T + b1[i]
    #pragma unroll
    for (int nt = 0; nt < 8; ++nt) {
      const int n = 16 * nt + lm;
      const float bia = b1[i * HID + n];
      v8f cc = acc[nt];
      #pragma unroll
      for (int j = 0; j < 8; ++j) cc[j] += bia;
      cc = wmma16x16x32(a0, *(const v16h*)(W1i + n * HID +  0 + 16 * hs), cc);
      cc = wmma16x16x32(a1, *(const v16h*)(W1i + n * HID + 32 + 16 * hs), cc);
      cc = wmma16x16x32(a2, *(const v16h*)(W1i + n * HID + 64 + 16 * hs), cc);
      cc = wmma16x16x32(a3, *(const v16h*)(W1i + n * HID + 96 + 16 * hs), cc);
      acc[nt] = cc;
    }
  }

  // ----------------- fc_out : leaky_relu(net) @ wout^T + bout --------------
  float p[8] = {0.f, 0.f, 0.f, 0.f, 0.f, 0.f, 0.f, 0.f};
  #pragma unroll
  for (int nt = 0; nt < 8; ++nt) {
    const float w = wout[16 * nt + lm];
    #pragma unroll
    for (int r = 0; r < 8; ++r) {
      const float x = acc[nt][r];
      p[r] += w * fmaxf(x, 0.2f * x);      // leaky_relu(x, 0.2)
    }
  }
  // reduce over the 16 lanes of each half-group (n dimension)
  #pragma unroll
  for (int r = 0; r < 8; ++r) {
    p[r] += __shfl_xor(p[r], 1, 32);
    p[r] += __shfl_xor(p[r], 2, 32);
    p[r] += __shfl_xor(p[r], 4, 32);
    p[r] += __shfl_xor(p[r], 8, 32);
  }
  if (lm == 0) {
    const float bo = bout[0];
    #pragma unroll
    for (int r = 0; r < 8; ++r) {
      const int qo = tile * 16 + r + 8 * hs;
      if (qo < nq) out[qo] = p[r] + bo;
    }
  }
}

// ---------------------------------------------------------------------------
extern "C" void kernel_launch(void* const* d_in, const int* in_sizes, int n_in,
                              void* d_out, int out_size, void* d_ws, size_t ws_size,
                              hipStream_t stream) {
  const float* zf      = (const float*)d_in[0];
  const float* qry     = (const float*)d_in[1];
  const float* fc_p_w  = (const float*)d_in[2];
  const float* fc_p_b  = (const float*)d_in[3];
  const float* fc_c_w  = (const float*)d_in[4];
  const float* fc_c_b  = (const float*)d_in[5];
  const float* fc0_w   = (const float*)d_in[6];
  const float* fc0_b   = (const float*)d_in[7];
  const float* fc1_w   = (const float*)d_in[8];
  const float* fc1_b   = (const float*)d_in[9];
  const float* fcout_w = (const float*)d_in[10];
  const float* fcout_b = (const float*)d_in[11];
  const int*   gsize   = (const int*)d_in[12];
  float* out = (float*)d_out;

  const int nq = in_sizes[1] / 3;

  // f16 weight cache in workspace, [n][k] row-major == WMMA B-fragment order
  _Float16* ws = (_Float16*)d_ws;
  _Float16* Wp = ws;                         // 128*32
  _Float16* Wc = ws + 4096;                  // 5*128*32
  _Float16* W0 = ws + 4096 + 20480;          // 5*128*128
  _Float16* W1 = ws + 4096 + 20480 + 81920;  // 5*128*128

  cvt_f32_f16_kernel<<<(4096  + 255) / 256, 256, 0, stream>>>(fc_p_w, Wp, 4096);
  cvt_f32_f16_kernel<<<(20480 + 255) / 256, 256, 0, stream>>>(fc_c_w, Wc, 20480);
  cvt_f32_f16_kernel<<<(81920 + 255) / 256, 256, 0, stream>>>(fc0_w,  W0, 81920);
  cvt_f32_f16_kernel<<<(81920 + 255) / 256, 256, 0, stream>>>(fc1_w,  W1, 81920);

  const int ntiles = (nq + 15) / 16;   // 16 queries per wave
  const int blocks = (ntiles + 7) / 8; // 8 waves per block
  PatchDecoder_84318797955409_kernel<<<blocks, 256, 0, stream>>>(
      zf, qry, Wp, Wc, W0, W1, fc_p_b, fc_c_b, fc0_b, fc1_b,
      fcout_w, fcout_b, out, nq, gsize);
}
